// WanSelfAttention_Sparse_7593502179443
// MI455X (gfx1250) — matrix operand order repaired and checked by reading, not compiled, not run on hardware
//
#include <hip/hip_runtime.h>
#include <hip/hip_bf16.h>

#define S_LEN 3072
#define MODEL_DIM 1536
#define HD 128
#define NH 12

typedef __attribute__((ext_vector_type(16))) _Float16 v16h;
typedef __attribute__((ext_vector_type(8)))  float    v8f;

union Frag16 { v16h v; uint4 q[2]; };

__device__ __forceinline__ v8f wmma16(const Frag16& a, const Frag16& b, v8f c) {
  // D(16x16,f32) = A(16x32,f16) * B(32x16,f16) + C
  return __builtin_amdgcn_wmma_f32_16x16x32_f16(false, a.v, false, b.v, (short)0, c,
                                                false, false);
}

// CDNA5 async global -> LDS copy (16 bytes), tracked by ASYNCcnt.
__device__ __forceinline__ void async_copy_b128(const _Float16* gptr, _Float16* lptr) {
  unsigned lds_addr = (unsigned)(size_t)lptr;             // low 32 bits = LDS offset
  unsigned long long gaddr = (unsigned long long)(size_t)gptr;
  asm volatile("global_load_async_to_lds_b128 %0, %1, off"
               :: "v"(lds_addr), "v"(gaddr) : "memory");
}

// ---------------------------------------------------------------------------
// f32 -> f16 conversion, 4 elements per thread
// ---------------------------------------------------------------------------
__global__ void cvt_f32_to_f16(const float* __restrict__ in,
                               _Float16* __restrict__ out, int n4) {
  int i = blockIdx.x * blockDim.x + threadIdx.x;
  if (i < n4) {
    float4 f = ((const float4*)in)[i];
    union { _Float16 h[4]; uint2 u; } pk;
    pk.h[0] = (_Float16)f.x; pk.h[1] = (_Float16)f.y;
    pk.h[2] = (_Float16)f.z; pk.h[3] = (_Float16)f.w;
    ((uint2*)out)[i] = pk.u;
  }
}

// ---------------------------------------------------------------------------
// WMMA GEMM:  C[M,N] = A[M,K] * B[N,K]^T + bias[N]
// A row-major (K contiguous, f16), B row-major [N,K] (f16).
// mode 0: write f32 row-major to Cf
// mode 1: write f16 transposed per head to Ch: (m, n) -> Ch[n/HD][n%HD][m]
// Block: 256 thr (8 waves), tile 128x128, K-step 32. Wave: 32 rows x 64 cols.
// Staging: double-buffered LDS fed by global_load_async_to_lds_b128,
// steady-state loop is branch-free (last tile peeled).
// ---------------------------------------------------------------------------
__launch_bounds__(256)
__global__ void gemm_f16(const _Float16* __restrict__ A,
                         const _Float16* __restrict__ B,
                         const float* __restrict__ bias,
                         float* __restrict__ Cf,
                         _Float16* __restrict__ Ch,
                         int M, int N, int K, int mode) {
  constexpr int LDT = 40;  // padded halfs per LDS tile row (80B, 16B aligned)
  __shared__ _Float16 Asm[2][128 * LDT];
  __shared__ _Float16 Bsm[2][128 * LDT];

  int tid = threadIdx.x;
  int lane = tid & 31;
  int w = tid >> 5;
  int wm = w >> 1;                // 0..3 -> 32-row strip
  int wn = w & 1;                 // 0..1 -> 64-col strip
  int bm = blockIdx.x * 128;
  int bn = blockIdx.y * 128;

  // Per-thread staging assignment: chunks c = tid and c = tid + 256.
  // chunk c covers tile row (c>>2), halfs [(c&3)*8, +8).  (tid+256)>>2 == r0+64.
  int r0 = tid >> 2;
  int kc = (tid & 3) * 8;
  const _Float16* aSrc0 = A + (size_t)(bm + r0) * K + kc;
  const _Float16* aSrc1 = A + (size_t)(bm + r0 + 64) * K + kc;
  const _Float16* bSrc0 = B + (size_t)(bn + r0) * K + kc;
  const _Float16* bSrc1 = B + (size_t)(bn + r0 + 64) * K + kc;
  int ldsOff = r0 * LDT + kc;

  v8f zero = {0.f, 0.f, 0.f, 0.f, 0.f, 0.f, 0.f, 0.f};
  v8f acc[2][4];
  for (int i = 0; i < 2; ++i)
    for (int j = 0; j < 4; ++j) acc[i][j] = zero;

  // fragment source addresses (constant per thread, swap buffers by index)
  int arow = wm * 32 + (lane & 15);
  int akoff = (lane >> 4) * 8;
  int bcol = wn * 64 + (lane & 15);
  int bkoff = (lane >> 4) * 8;

  auto stage = [&](int buf, int koff) {
    async_copy_b128(aSrc0 + koff, &Asm[buf][ldsOff]);
    async_copy_b128(aSrc1 + koff, &Asm[buf][ldsOff + 64 * LDT]);
    async_copy_b128(bSrc0 + koff, &Bsm[buf][ldsOff]);
    async_copy_b128(bSrc1 + koff, &Bsm[buf][ldsOff + 64 * LDT]);
  };
  auto compute = [&](int buf) {
    Frag16 af[2], bf[4];
    for (int i = 0; i < 2; ++i) {
      const _Float16* p = &Asm[buf][(arow + i * 16) * LDT + akoff];
      af[i].q[0] = *(const uint4*)p;
      af[i].q[1] = *(const uint4*)(p + 16);
    }
    for (int j = 0; j < 4; ++j) {
      const _Float16* p = &Bsm[buf][(bcol + j * 16) * LDT + bkoff];
      bf[j].q[0] = *(const uint4*)p;
      bf[j].q[1] = *(const uint4*)(p + 16);
    }
    for (int i = 0; i < 2; ++i)
      for (int j = 0; j < 4; ++j)
        acc[i][j] = wmma16(af[i], bf[j], acc[i][j]);
  };

  // prime the pipeline: stage tile 0 into buffer 0
  stage(0, 0);

  int nk = K >> 5;
  // steady state: branch-free body; stage t+1, wait for t, compute t
  for (int t = 0; t < nk - 1; ++t) {
    int cur = t & 1;
    stage(1 - cur, (t + 1) * 32);
    // 8 outstanding; in-order completion: <=4 left means tile t has landed
    asm volatile("s_wait_asynccnt 0x4" ::: "memory");
    __syncthreads();
    compute(cur);
    // all reads of buf[cur] must complete before it is re-staged next iter
    asm volatile("s_wait_dscnt 0x0" ::: "memory");
    __syncthreads();
  }
  // peeled last tile
  asm volatile("s_wait_asynccnt 0x0" ::: "memory");
  __syncthreads();
  compute((nk - 1) & 1);

  // epilogue (D layout: lane&15 = col, (lane>>4)*8 + r = row)
  int colb = lane & 15;
  int rowb = (lane >> 4) * 8;
  for (int i = 0; i < 2; ++i) {
    for (int j = 0; j < 4; ++j) {
      int gn = bn + wn * 64 + j * 16 + colb;
      float bsv = bias ? bias[gn] : 0.f;
      for (int r = 0; r < 8; ++r) {
        int gm = bm + wm * 32 + i * 16 + rowb + r;
        float val = acc[i][j][r] + bsv;
        if (mode == 0) {
          Cf[(size_t)gm * N + gn] = val;
        } else {
          int h = gn >> 7, jd = gn & 127;
          Ch[(size_t)h * HD * M + (size_t)jd * M + gm] = (_Float16)val;
        }
      }
    }
  }
}

// ---------------------------------------------------------------------------
// per-token inverse RMS of q_raw (y=0) / k_raw (y=1)
// ---------------------------------------------------------------------------
__global__ void rms_rows(const float* __restrict__ q, const float* __restrict__ k,
                         float* __restrict__ rq, float* __restrict__ rk) {
  int m = blockIdx.x;
  const float* src = (blockIdx.y == 0) ? q : k;
  float* dst = (blockIdx.y == 0) ? rq : rk;
  float s = 0.f;
  for (int i = threadIdx.x; i < MODEL_DIM; i += 256) {
    float v = src[(size_t)m * MODEL_DIM + i];
    s += v * v;
  }
  __shared__ float red[256];
  red[threadIdx.x] = s;
  __syncthreads();
  for (int o = 128; o > 0; o >>= 1) {
    if (threadIdx.x < o) red[threadIdx.x] += red[threadIdx.x + o];
    __syncthreads();
  }
  if (threadIdx.x == 0)
    dst[m] = rsqrtf(red[0] / (float)MODEL_DIM + 1e-6f);
}

// ---------------------------------------------------------------------------
// RMSNorm-apply + gamma + 3D RoPE + pack to f16 head layout [h][s][d]
// grid (s, NH), block 64 (one complex pair per thread)
// ---------------------------------------------------------------------------
__global__ void rope_pack(const float* __restrict__ qraw, const float* __restrict__ kraw,
                          const float* __restrict__ rq, const float* __restrict__ rk,
                          const float* __restrict__ gq, const float* __restrict__ gk,
                          const float* __restrict__ freqs, const int* __restrict__ gs,
                          _Float16* __restrict__ qh, _Float16* __restrict__ kh) {
  int m = blockIdx.x, h = blockIdx.y, p = threadIdx.x;  // p in [0,64)
  int Hs = gs[1], Ws = gs[2];
  int fi = m / (Hs * Ws);
  int hi = (m / Ws) % Hs;
  int wi = m % Ws;
  int pos = (p < 22) ? fi : ((p < 43) ? hi : wi);
  float ang = freqs[pos * 64 + p];
  float sn, cs;
  __sincosf(ang, &sn, &cs);
  float riq = rq[m], rik = rk[m];
  int c0 = h * HD + 2 * p;
  size_t src = (size_t)m * MODEL_DIM + c0;
  float qa = qraw[src] * riq * gq[c0];
  float qb = qraw[src + 1] * riq * gq[c0 + 1];
  float ka = kraw[src] * rik * gk[c0];
  float kb = kraw[src + 1] * rik * gk[c0 + 1];
  size_t o = (size_t)h * S_LEN * HD + (size_t)m * HD + 2 * p;
  qh[o]     = (_Float16)(qa * cs - qb * sn);
  qh[o + 1] = (_Float16)(qa * sn + qb * cs);
  kh[o]     = (_Float16)(ka * cs - kb * sn);
  kh[o + 1] = (_Float16)(ka * sn + kb * cs);
}

// ---------------------------------------------------------------------------
// Flash attention. grid (S/128, NH), block 256 (8 waves).
// Each wave owns a 16-row query tile, iterates keys in steps of 32.
// qh,kh: [NH][S][HD] f16 ; vt: [NH][HD][S] f16 ; oh: [S][MODEL_DIM] f16
// ---------------------------------------------------------------------------
__launch_bounds__(256)
__global__ void flash_attn(const _Float16* __restrict__ qh,
                           const _Float16* __restrict__ kh,
                           const _Float16* __restrict__ vt,
                           _Float16* __restrict__ oh) {
  __shared__ _Float16 Plds[8][16 * 40];  // per-wave P tile (padded, 16B aligned)
  int tid = threadIdx.x, lane = tid & 31, w = tid >> 5;
  int h = blockIdx.y;
  int q0 = blockIdx.x * 128 + w * 16;
  const _Float16* Q = qh + (size_t)h * S_LEN * HD;
  const _Float16* Kp = kh + (size_t)h * S_LEN * HD;
  const _Float16* Vp = vt + (size_t)h * HD * S_LEN;
  _Float16* Pw = &Plds[w][0];

  // Q A-fragments for the 4 K=32 chunks of d=128 (kept in registers)
  Frag16 aq[4];
  {
    int row = q0 + (lane & 15);
    int koff = (lane >> 4) * 8;
    for (int c = 0; c < 4; ++c) {
      const _Float16* p = Q + (size_t)row * HD + c * 32 + koff;
      aq[c].q[0] = *(const uint4*)p;
      aq[c].q[1] = *(const uint4*)(p + 16);
    }
  }

  v8f zero = {0.f, 0.f, 0.f, 0.f, 0.f, 0.f, 0.f, 0.f};
  v8f acc[8];
  for (int t = 0; t < 8; ++t) acc[t] = zero;
  float mst[8], lst[8];
  for (int r = 0; r < 8; ++r) { mst[r] = -1e30f; lst[r] = 0.f; }
  const float scale = 0.08838834764831845f;  // 1/sqrt(128)

  int bcol = lane & 15;
  int bkoff = (lane >> 4) * 8;

  for (int t0 = 0; t0 < S_LEN; t0 += 32) {
    // S = Q * K^T for 32 keys (two 16x16 D tiles)
    v8f s0 = zero, s1 = zero;
    for (int c = 0; c < 4; ++c) {
      Frag16 bk0, bk1;
      const _Float16* p0 = Kp + (size_t)(t0 + bcol) * HD + c * 32 + bkoff;
      bk0.q[0] = *(const uint4*)p0;
      bk0.q[1] = *(const uint4*)(p0 + 16);
      const _Float16* p1 = Kp + (size_t)(t0 + 16 + bcol) * HD + c * 32 + bkoff;
      bk1.q[0] = *(const uint4*)p1;
      bk1.q[1] = *(const uint4*)(p1 + 16);
      s0 = wmma16(aq[c], bk0, s0);
      s1 = wmma16(aq[c], bk1, s1);
    }

    // online softmax: rows live in lane halves; xor masks 1..8 stay in-half
    float e0[8], e1[8], rmax[8], al[8], rs[8];
    for (int r = 0; r < 8; ++r) {
      e0[r] = s0[r] * scale;
      e1[r] = s1[r] * scale;
      rmax[r] = fmaxf(e0[r], e1[r]);
    }
    for (int msk = 1; msk <= 8; msk <<= 1)
      for (int r = 0; r < 8; ++r)
        rmax[r] = fmaxf(rmax[r], __shfl_xor(rmax[r], msk, 32));
    for (int r = 0; r < 8; ++r) {
      float mn = fmaxf(mst[r], rmax[r]);
      al[r] = __expf(mst[r] - mn);
      e0[r] = __expf(e0[r] - mn);
      e1[r] = __expf(e1[r] - mn);
      rs[r] = e0[r] + e1[r];
      mst[r] = mn;
    }
    for (int msk = 1; msk <= 8; msk <<= 1)
      for (int r = 0; r < 8; ++r)
        rs[r] += __shfl_xor(rs[r], msk, 32);
    for (int r = 0; r < 8; ++r) lst[r] = lst[r] * al[r] + rs[r];
    for (int t = 0; t < 8; ++t)
      for (int r = 0; r < 8; ++r) acc[t][r] *= al[r];

    // P (D layout) -> LDS -> A-fragment layout
    int prow = (lane >> 4) * 8;
    for (int r = 0; r < 8; ++r) {
      Pw[(prow + r) * 40 + bcol] = (_Float16)e0[r];
      Pw[(prow + r) * 40 + 16 + bcol] = (_Float16)e1[r];
    }
    asm volatile("s_wait_dscnt 0x0" ::: "memory");
    Frag16 pf;
    {
      int row = lane & 15;
      int koff = (lane >> 4) * 8;
      const _Float16* p = Pw + row * 40 + koff;
      pf.q[0] = *(const uint4*)p;
      pf.q[1] = *(const uint4*)(p + 16);
    }

    // O += P * V  (V transposed per head: [dcol][key], key contiguous)
    for (int t = 0; t < 8; ++t) {
      Frag16 bv;
      const _Float16* p = Vp + (size_t)(t * 16 + bcol) * S_LEN + t0 + bkoff;
      bv.q[0] = *(const uint4*)p;
      bv.q[1] = *(const uint4*)(p + 16);
      acc[t] = wmma16(pf, bv, acc[t]);
    }
  }

  // finalize: divide by l, store f16 to [S][MODEL_DIM]
  int rowb = (lane >> 4) * 8;
  float inv[8];
  for (int r = 0; r < 8; ++r) inv[r] = 1.0f / lst[r];
  for (int t = 0; t < 8; ++t)
    for (int r = 0; r < 8; ++r) {
      float v = acc[t][r] * inv[r];
      oh[(size_t)(q0 + rowb + r) * MODEL_DIM + h * HD + t * 16 + bcol] = (_Float16)v;
    }
}

// ---------------------------------------------------------------------------
extern "C" void kernel_launch(void* const* d_in, const int* in_sizes, int n_in,
                              void* d_out, int out_size, void* d_ws, size_t ws_size,
                              hipStream_t stream) {
  const float* x = (const float*)d_in[0];
  const int* grid_sizes = (const int*)d_in[2];
  const float* freqs = (const float*)d_in[3];
  const float* Wq = (const float*)d_in[5];
  const float* bq = (const float*)d_in[6];
  const float* Wk = (const float*)d_in[7];
  const float* bk = (const float*)d_in[8];
  const float* Wv = (const float*)d_in[9];
  const float* bv = (const float*)d_in[10];
  const float* Wo = (const float*)d_in[11];
  const float* bo = (const float*)d_in[12];
  const float* gq = (const float*)d_in[13];
  const float* gk = (const float*)d_in[14];
  float* out = (float*)d_out;

  char* ws = (char*)d_ws;
  size_t off = 0;
  auto alloc = [&](size_t bytes) -> void* {
    void* p = ws + off;
    off += (bytes + 255) & ~(size_t)255;
    return p;
  };
  const size_t SD = (size_t)S_LEN * MODEL_DIM;   // 4.7M elems
  const size_t DD = (size_t)MODEL_DIM * MODEL_DIM;

  _Float16* xh  = (_Float16*)alloc(SD * 2);
  _Float16* wqh = (_Float16*)alloc(DD * 2);
  _Float16* wkh = (_Float16*)alloc(DD * 2);
  _Float16* wvh = (_Float16*)alloc(DD * 2);
  _Float16* woh = (_Float16*)alloc(DD * 2);
  float* qraw   = (float*)alloc(SD * 4);
  float* kraw   = (float*)alloc(SD * 4);
  _Float16* qh  = (_Float16*)alloc(SD * 2);
  _Float16* khd = (_Float16*)alloc(SD * 2);
  _Float16* vt  = (_Float16*)alloc(SD * 2);
  _Float16* oh  = (_Float16*)alloc(SD * 2);
  float* rq     = (float*)alloc(S_LEN * 4);
  float* rk     = (float*)alloc(S_LEN * 4);
  (void)ws_size; (void)in_sizes; (void)n_in; (void)out_size;

  // 1) f32 -> f16 conversions
  {
    int n4 = (int)(SD / 4);
    cvt_f32_to_f16<<<(n4 + 255) / 256, 256, 0, stream>>>(x, xh, n4);
    int w4 = (int)(DD / 4);
    int g = (w4 + 255) / 256;
    cvt_f32_to_f16<<<g, 256, 0, stream>>>(Wq, wqh, w4);
    cvt_f32_to_f16<<<g, 256, 0, stream>>>(Wk, wkh, w4);
    cvt_f32_to_f16<<<g, 256, 0, stream>>>(Wv, wvh, w4);
    cvt_f32_to_f16<<<g, 256, 0, stream>>>(Wo, woh, w4);
  }

  dim3 ggrid(S_LEN / 128, MODEL_DIM / 128);
  // 2) Q,K raw projections (f32), V projection direct to per-head transposed f16
  gemm_f16<<<ggrid, 256, 0, stream>>>(xh, wqh, bq, qraw, nullptr,
                                      S_LEN, MODEL_DIM, MODEL_DIM, 0);
  gemm_f16<<<ggrid, 256, 0, stream>>>(xh, wkh, bk, kraw, nullptr,
                                      S_LEN, MODEL_DIM, MODEL_DIM, 0);
  gemm_f16<<<ggrid, 256, 0, stream>>>(xh, wvh, bv, nullptr, vt,
                                      S_LEN, MODEL_DIM, MODEL_DIM, 1);

  // 3) RMS stats + norm/RoPE/pack
  rms_rows<<<dim3(S_LEN, 2), 256, 0, stream>>>(qraw, kraw, rq, rk);
  rope_pack<<<dim3(S_LEN, NH), 64, 0, stream>>>(qraw, kraw, rq, rk, gq, gk,
                                                freqs, grid_sizes, qh, khd);

  // 4) attention
  flash_attn<<<dim3(S_LEN / 128, NH), 256, 0, stream>>>(qh, khd, vt, oh);

  // 5) output projection -> f32 out
  gemm_f16<<<ggrid, 256, 0, stream>>>(oh, woh, bo, out, nullptr,
                                      S_LEN, MODEL_DIM, MODEL_DIM, 0);
}
